// TriangleAttention_63788854280317
// MI455X (gfx1250) — compile-verified, matrix operand order checked by reading, not run on hardware
//
#include <hip/hip_runtime.h>
#include <hip/hip_bf16.h>

typedef __bf16 bf16_t;
typedef __attribute__((ext_vector_type(16))) __bf16 v16bf;
typedef __attribute__((ext_vector_type(8)))  float  v8f;

#define LSEQ  256
#define CDIM  256
#define HEADS 8
#define DHEAD 32
#define ATTN_SCALE 0.17677669529663687f   // 32^-0.5

// ---------------------------------------------------------------------------
// Fragment loader for CDNA5 16-bit WMMA operands with contiguous-K memory.
// A-matrix (16xK): row M = lane%16, K in halves. B-matrix (Kx16): col N =
// lane%16, K in halves. Both use: halves 0..7 = K {0..7 | 8..15} (by lane
// group), halves 8..15 = K {16..23 | 24..31}.  base points at (r=0, k=k0),
// ld = elements between rows/cols.  Requires 16B alignment (ld % 8 == 0).
// ---------------------------------------------------------------------------
__device__ __forceinline__ v16bf load_frag_k(const bf16_t* base, int ld, int lane) {
  const int sel = (lane >> 4) & 1;
  const bf16_t* p = base + (lane & 15) * ld + sel * 8;
  union { v16bf v; uint4 u[2]; } f;
  f.u[0] = *reinterpret_cast<const uint4*>(p);        // K = sel*8 .. sel*8+7
  f.u[1] = *reinterpret_cast<const uint4*>(p + 16);   // K = 16+sel*8 .. +7
  return f.v;
}

__device__ __forceinline__ v8f wmma_bf16(v16bf a, v16bf b, v8f c) {
  return __builtin_amdgcn_wmma_f32_16x16x32_bf16(false, a, false, b,
                                                 (short)0, c, false, false);
}

// ---------------------------------------------------------------------------
// fp32 -> bf16 weight conversion
// ---------------------------------------------------------------------------
__global__ void cvt_bf16_kernel(const float* __restrict__ w,
                                bf16_t* __restrict__ o, int n) {
  int i = blockIdx.x * 256 + threadIdx.x;
  if (i < n) o[i] = (bf16_t)w[i];
}

// ---------------------------------------------------------------------------
// LayerNorm over last dim (C=256) + cast to bf16.  1 wave per row, 8 rows per
// 256-thread block.  65536 rows total.
// ---------------------------------------------------------------------------
__global__ void ln_kernel(const float* __restrict__ x,
                          const float* __restrict__ gamma,
                          const float* __restrict__ beta,
                          bf16_t* __restrict__ xo) {
  const int wave = threadIdx.x >> 5;
  const int lane = threadIdx.x & 31;
  const long row = (long)blockIdx.x * 8 + wave;
  const float* xr = x + row * CDIM;

  float vals[8];
  float s = 0.f;
#pragma unroll
  for (int t = 0; t < 8; ++t) { vals[t] = xr[lane + t * 32]; s += vals[t]; }
#pragma unroll
  for (int m = 16; m; m >>= 1) s += __shfl_xor(s, m, 32);
  const float mu = s * (1.f / 256.f);

  float vs = 0.f;
#pragma unroll
  for (int t = 0; t < 8; ++t) { float d = vals[t] - mu; vs += d * d; }
#pragma unroll
  for (int m = 16; m; m >>= 1) vs += __shfl_xor(vs, m, 32);
  const float rstd = rsqrtf(vs * (1.f / 256.f) + 1e-5f);

  bf16_t* orow = xo + row * CDIM;
#pragma unroll
  for (int t = 0; t < 8; ++t) {
    int c = lane + t * 32;
    orow[c] = (bf16_t)((vals[t] - mu) * rstd * gamma[c] + beta[c]);
  }
}

// ---------------------------------------------------------------------------
// QKV projection: (65536 x 256) @ (256 x 256)^T for q/k/v simultaneously
// (A-fragment reused across 3 WMMAs).  Results scattered (with bias) to the
// attention-friendly layout  [j][h][i][d]  as bf16.
// grid = (512, 16), block = 256 (8 waves); wave tile = 16x16, K-loop of 8.
// ---------------------------------------------------------------------------
__global__ void qkv_kernel(const bf16_t* __restrict__ xb,
                           const bf16_t* __restrict__ wq,
                           const bf16_t* __restrict__ wk,
                           const bf16_t* __restrict__ wv,
                           const float* __restrict__ bq,
                           const float* __restrict__ bk,
                           const float* __restrict__ bv,
                           bf16_t* __restrict__ qo,
                           bf16_t* __restrict__ ko,
                           bf16_t* __restrict__ vo) {
  const int lane = threadIdx.x & 31;
  const int wave = threadIdx.x >> 5;
  const long m0 = ((long)blockIdx.x * 8 + wave) * 16;
  const int  n0 = blockIdx.y * 16;

  v8f accq = {}, acck = {}, accv = {};
#pragma unroll
  for (int kt = 0; kt < CDIM; kt += 32) {
    v16bf a  = load_frag_k(xb + m0 * CDIM + kt, CDIM, lane);
    v16bf fq = load_frag_k(wq + n0 * CDIM + kt, CDIM, lane);
    v16bf fk = load_frag_k(wk + n0 * CDIM + kt, CDIM, lane);
    v16bf fv = load_frag_k(wv + n0 * CDIM + kt, CDIM, lane);
    accq = wmma_bf16(a, fq, accq);
    acck = wmma_bf16(a, fk, acck);
    accv = wmma_bf16(a, fv, accv);
  }

  const int r = lane & 15, sel = lane >> 4;
  const int n = n0 + r;
  const int h = n >> 5, d = n & 31;
  const float vbq = bq[n], vbk = bk[n], vbv = bv[n];
#pragma unroll
  for (int g = 0; g < 8; ++g) {
    long m = m0 + g + sel * 8;
    int i = (int)(m >> 8), j = (int)(m & 255);
    long o = (((long)j * HEADS + h) * LSEQ + i) * DHEAD + d;
    qo[o] = (bf16_t)(accq[g] + vbq);
    ko[o] = (bf16_t)(acck[g] + vbk);
    vo[o] = (bf16_t)(accv[g] + vbv);
  }
}

// ---------------------------------------------------------------------------
// Attention per (j,h): logits = Q K^T * scale (masked), softmax over k,
// attn = P V.  1 block per (j,h) (2048 blocks), 128 threads (4 waves), each
// wave handles 4 i-tiles of 16 rows.  K staged into LDS via the CDNA5 async
// memory->LDS DMA path (GLOBAL_LOAD_ASYNC_TO_LDS_B128, ASYNCcnt); V staged
// transposed; probs staged per-wave in LDS to feed the PV A-fragments.
// Dynamic LDS: 16KB (K) + 16KB (V^T) + 4*8KB (probs) = 64KB.
// ---------------------------------------------------------------------------
__global__ void attn_kernel(const bf16_t* __restrict__ qg,
                            const bf16_t* __restrict__ kg,
                            const bf16_t* __restrict__ vg,
                            const int* __restrict__ mask,
                            bf16_t* __restrict__ attn) {
  extern __shared__ char smem[];
  bf16_t* Ksh  = (bf16_t*)smem;                 // [k][d]   256*32
  bf16_t* VTsh = Ksh + LSEQ * DHEAD;            // [d][k]   32*256
  bf16_t* Psh  = VTsh + DHEAD * LSEQ;           // per-wave [16][256]

  const int lane = threadIdx.x & 31;
  const int wave = threadIdx.x >> 5;            // 0..3
  const int jh = blockIdx.x;
  const int j = jh >> 3, h = jh & 7;

  const bf16_t* Kg = kg + (long)jh * LSEQ * DHEAD;
  const bf16_t* Vg = vg + (long)jh * LSEQ * DHEAD;
  const bf16_t* Qg = qg + (long)jh * LSEQ * DHEAD;

  // ---- Stage K via async DMA (global -> LDS, no VGPR round trip) ----
  {
    const unsigned ldsK = (unsigned)(uintptr_t)(void*)Ksh;
    for (int idx = threadIdx.x; idx < LSEQ * DHEAD / 8; idx += 128) {
      unsigned long long ga =
          (unsigned long long)(uintptr_t)((const char*)Kg + idx * 16);
      unsigned la = ldsK + idx * 16;
      asm volatile("global_load_async_to_lds_b128 %0, %1, off"
                   :: "v"(la), "v"(ga) : "memory");
    }
    asm volatile("s_wait_asynccnt 0x0" ::: "memory");
  }

  // ---- Stage V transposed into LDS ----
  for (int idx = threadIdx.x; idx < LSEQ * DHEAD; idx += 128) {
    int kk = idx >> 5, d = idx & 31;
    VTsh[d * LSEQ + kk] = Vg[idx];
  }
  __syncthreads();

  bf16_t* Pw = Psh + wave * 16 * LSEQ;
  const int r = lane & 15, sel = lane >> 4;

  for (int it = 0; it < 4; ++it) {
    const int i0 = (it * 4 + wave) * 16;

    // ---- logits: Q(16x32) x K^T(32x16) per k-tile, 16 tiles ----
    v16bf aq = load_frag_k(Qg + i0 * DHEAD, DHEAD, lane);
    v8f acc[16];
#pragma unroll
    for (int t = 0; t < 16; ++t) {
      v16bf fb = load_frag_k(Ksh + (t * 16) * DHEAD, DHEAD, lane);
      v8f c = {};
      acc[t] = wmma_bf16(aq, fb, c);
    }

    // ---- masked softmax over k (per row i = i0 + g + 8*sel) ----
#pragma unroll
    for (int g = 0; g < 8; ++g) {
      const int i = i0 + g + sel * 8;
      float mx = -INFINITY;
#pragma unroll
      for (int t = 0; t < 16; ++t) {
        int kcol = t * 16 + r;
        float lg = acc[t][g] * ATTN_SCALE;
        lg = (mask[i * LSEQ + kcol] != 0) ? lg : -INFINITY;
        acc[t][g] = lg;
        mx = fmaxf(mx, lg);
      }
#pragma unroll
      for (int mm = 8; mm; mm >>= 1) mx = fmaxf(mx, __shfl_xor(mx, mm, 32));
      float sum = 0.f;
#pragma unroll
      for (int t = 0; t < 16; ++t) {
        float e = __expf(acc[t][g] - mx);
        acc[t][g] = e;
        sum += e;
      }
#pragma unroll
      for (int mm = 8; mm; mm >>= 1) sum += __shfl_xor(sum, mm, 32);
      const float inv = 1.f / sum;
#pragma unroll
      for (int t = 0; t < 16; ++t)
        Pw[(g + sel * 8) * LSEQ + t * 16 + r] = (bf16_t)(acc[t][g] * inv);
    }

    // ---- attn = P(16x256) x V(256x32): K-loop of 8, two 16-wide d halves ---
    v8f oacc0 = {}, oacc1 = {};
#pragma unroll
    for (int kc = 0; kc < LSEQ; kc += 32) {
      v16bf ap = load_frag_k(Pw + kc, LSEQ, lane);
      v16bf b0 = load_frag_k(VTsh + kc, LSEQ, lane);
      v16bf b1 = load_frag_k(VTsh + 16 * LSEQ + kc, LSEQ, lane);
      oacc0 = wmma_bf16(ap, b0, oacc0);
      oacc1 = wmma_bf16(ap, b1, oacc1);
    }

    // ---- store attn[(i*L + j)*C + h*32 + d] as bf16 ----
#pragma unroll
    for (int g = 0; g < 8; ++g) {
      const int i = i0 + g + sel * 8;
      long base = ((long)i * LSEQ + j) * CDIM + h * DHEAD;
      attn[base + r]      = (bf16_t)oacc0[g];
      attn[base + 16 + r] = (bf16_t)oacc1[g];
    }
  }
}

// ---------------------------------------------------------------------------
// Output projection + bias + residual: out = attn @ Wo^T + bo + pair  (fp32)
// grid = (512, 16), block = 256 (8 waves); wave tile = 16x16.
// ---------------------------------------------------------------------------
__global__ void out_kernel(const bf16_t* __restrict__ attn,
                           const bf16_t* __restrict__ wo,
                           const float* __restrict__ bo,
                           const float* __restrict__ pair,
                           float* __restrict__ out) {
  const int lane = threadIdx.x & 31;
  const int wave = threadIdx.x >> 5;
  const long m0 = ((long)blockIdx.x * 8 + wave) * 16;
  const int  n0 = blockIdx.y * 16;

  v8f acc = {};
#pragma unroll
  for (int kt = 0; kt < CDIM; kt += 32) {
    v16bf a = load_frag_k(attn + m0 * CDIM + kt, CDIM, lane);
    v16bf b = load_frag_k(wo + n0 * CDIM + kt, CDIM, lane);
    acc = wmma_bf16(a, b, acc);
  }

  const int r = lane & 15, sel = lane >> 4;
  const int n = n0 + r;
  const float bn = bo[n];
#pragma unroll
  for (int g = 0; g < 8; ++g) {
    long m = m0 + g + sel * 8;
    out[m * CDIM + n] = acc[g] + bn + pair[m * CDIM + n];
  }
}

// ---------------------------------------------------------------------------
extern "C" void kernel_launch(void* const* d_in, const int* in_sizes, int n_in,
                              void* d_out, int out_size, void* d_ws, size_t ws_size,
                              hipStream_t stream) {
  const float* pair  = (const float*)d_in[0];
  const int*   mask  = (const int*)  d_in[1];
  const float* gamma = (const float*)d_in[2];
  const float* beta  = (const float*)d_in[3];
  const float* Wq    = (const float*)d_in[4];
  const float* bq    = (const float*)d_in[5];
  const float* Wk    = (const float*)d_in[6];
  const float* bk    = (const float*)d_in[7];
  const float* Wv    = (const float*)d_in[8];
  const float* bv    = (const float*)d_in[9];
  const float* Wo    = (const float*)d_in[10];
  const float* bo    = (const float*)d_in[11];
  float* out = (float*)d_out;

  char* ws = (char*)d_ws;
  const size_t ROWB = (size_t)LSEQ * LSEQ * CDIM * sizeof(bf16_t);  // 33.5 MB
  bf16_t* xb  = (bf16_t*)(ws);            // layernormed x; reused for attn out
  bf16_t* qb  = (bf16_t*)(ws + 1 * ROWB);
  bf16_t* kb  = (bf16_t*)(ws + 2 * ROWB);
  bf16_t* vb  = (bf16_t*)(ws + 3 * ROWB);
  bf16_t* wqb = (bf16_t*)(ws + 4 * ROWB);
  bf16_t* wkb = wqb + CDIM * CDIM;
  bf16_t* wvb = wkb + CDIM * CDIM;
  bf16_t* wob = wvb + CDIM * CDIM;

  cvt_bf16_kernel<<<256, 256, 0, stream>>>(Wq, wqb, CDIM * CDIM);
  cvt_bf16_kernel<<<256, 256, 0, stream>>>(Wk, wkb, CDIM * CDIM);
  cvt_bf16_kernel<<<256, 256, 0, stream>>>(Wv, wvb, CDIM * CDIM);
  cvt_bf16_kernel<<<256, 256, 0, stream>>>(Wo, wob, CDIM * CDIM);

  ln_kernel<<<LSEQ * LSEQ / 8, 256, 0, stream>>>(pair, gamma, beta, xb);

  qkv_kernel<<<dim3(LSEQ * LSEQ / 128, CDIM / 16), 256, 0, stream>>>(
      xb, wqb, wkb, wvb, bq, bk, bv, qb, kb, vb);

  attn_kernel<<<LSEQ * HEADS, 128, 65536, stream>>>(qb, kb, vb, mask, xb);

  out_kernel<<<dim3(LSEQ * LSEQ / 128, CDIM / 16), 256, 0, stream>>>(
      xb, wob, bo, pair, out);
}
